// GeniePathConv_36429912605267
// MI455X (gfx1250) — matrix operand order repaired and checked by reading, not compiled, not run on hardware
//
#include <hip/hip_runtime.h>
#include <math.h>

#define DHID 128
#define NSLOPE 0.2f

typedef __attribute__((ext_vector_type(16))) __bf16 v16bf;
typedef __attribute__((ext_vector_type(8)))  __bf16 v8bf;
typedef __attribute__((ext_vector_type(8)))  float  v8f;

// Native bf16 convert (RNE fptrunc -> v_cvt on gfx1250)
__device__ __forceinline__ __bf16 f2bf(float f) { return (__bf16)f; }

// Exact bf16 -> f32 (bit shift)
__device__ __forceinline__ float bfbits2f(unsigned u16) {
  return __uint_as_float(u16 << 16);
}

__device__ __forceinline__ float sigmoidf_(float x) { return 1.0f / (1.0f + __expf(-x)); }

__device__ __forceinline__ void atomicMaxFloat(float* addr, float val) {
  if (val >= 0.0f) atomicMax((int*)addr, __float_as_int(val));
  else             atomicMin((unsigned int*)addr, __float_as_uint(val));
}

// ---------------------------------------------------------------------------
// Weight pre-conversion: fp32 -> bf16; W additionally transposed so that a
// WMMA B-fragment (16 contiguous K values per lane) is a single 32B load.
// ---------------------------------------------------------------------------
__global__ __launch_bounds__(256) void conv_w_transpose(const float* __restrict__ W,
                                                        __bf16* __restrict__ Wt) {
  int i = blockIdx.x * 256 + threadIdx.x;      // over 128*128
  int k = i >> 7, n = i & 127;
  Wt[n * DHID + k] = f2bf(W[k * DHID + n]);
}

__global__ __launch_bounds__(256) void conv_bf16(const float* __restrict__ A,
                                                 __bf16* __restrict__ B, int total) {
  int i = blockIdx.x * 256 + threadIdx.x;
  if (i < total) B[i] = f2bf(A[i]);
}

// ---------------------------------------------------------------------------
// feat = X @ W  (N x 128) x (128 x 128), bf16 WMMA, fp32 accumulate.
// One wave computes a 16-row x 128-col stripe: 8 col-tiles x 4 K-steps = 32
// v_wmma_f32_16x16x32_bf16. A loaded from global fp32 (rows are wave-unique),
// B from the 32KB cached transposed bf16 weight. feat is stored as bf16:
// its only consumers (attention dots, edge gather) are bandwidth-bound.
// ---------------------------------------------------------------------------
__global__ __launch_bounds__(128) void gat_gemm_wmma(const float* __restrict__ X,
                                                     const __bf16* __restrict__ Bt,
                                                     __bf16* __restrict__ F, int nrows) {
  const int lane = threadIdx.x & 31;
  const int wave = threadIdx.x >> 5;
  const long long rowBase = ((long long)blockIdx.x * 4 + wave) * 16;
  if (rowBase + 16 > nrows) return;            // wave-uniform: EXEC stays all-1 for WMMA
  const int am = lane & 15;                    // A-row in tile / C column
  const int ah = lane >> 4;                    // lane-half selector

  v8f acc[8];
#pragma unroll
  for (int t = 0; t < 8; ++t)
#pragma unroll
    for (int g = 0; g < 8; ++g) acc[t][g] = 0.0f;

  const float* arow = X + (rowBase + am) * DHID;
#pragma unroll
  for (int kk = 0; kk < 4; ++kk) {
    // A fragment: half 0 holds K 0-15, half 1 holds K 16-31
    const float* ap = arow + kk * 32 + ah * 16;
    float4 f0 = *(const float4*)(ap);
    float4 f1 = *(const float4*)(ap + 4);
    float4 f2 = *(const float4*)(ap + 8);
    float4 f3 = *(const float4*)(ap + 12);
    v16bf a;
    a[0]=f2bf(f0.x); a[1]=f2bf(f0.y); a[2]=f2bf(f0.z); a[3]=f2bf(f0.w);
    a[4]=f2bf(f1.x); a[5]=f2bf(f1.y); a[6]=f2bf(f1.z); a[7]=f2bf(f1.w);
    a[8]=f2bf(f2.x); a[9]=f2bf(f2.y); a[10]=f2bf(f2.z); a[11]=f2bf(f2.w);
    a[12]=f2bf(f3.x); a[13]=f2bf(f3.y); a[14]=f2bf(f3.z); a[15]=f2bf(f3.w);
#pragma unroll
    for (int t = 0; t < 8; ++t) {
      const __bf16* bp = Bt + (t * 16 + am) * DHID + kk * 32 + ah * 16;
      v16bf b = *(const v16bf*)bp;             // 32B aligned contiguous K run
      acc[t] = __builtin_amdgcn_wmma_f32_16x16x32_bf16(
          false, a, false, b, (short)0, acc[t], false, false);
    }
  }
  // C layout: VGPR g -> row g (+8 for lane half 1), col = lane&15
#pragma unroll
  for (int t = 0; t < 8; ++t)
#pragma unroll
    for (int g = 0; g < 8; ++g)
      F[(rowBase + g + ah * 8) * DHID + t * 16 + am] = f2bf(acc[t][g]);
}

// ---------------------------------------------------------------------------
// el[n] = <feat[n], attn_l>, er[n] = <feat[n], attn_r>  (one wave per row,
// feat read as packed bf16: 8B per lane)
// ---------------------------------------------------------------------------
__global__ __launch_bounds__(256) void node_attn_scores(const __bf16* __restrict__ F,
                                                        const float* __restrict__ attn_l,
                                                        const float* __restrict__ attn_r,
                                                        float* __restrict__ el,
                                                        float* __restrict__ er, int nrows) {
  const long long row = (long long)blockIdx.x * 8 + (threadIdx.x >> 5);
  if (row >= nrows) return;
  const int lane = threadIdx.x & 31;
  uint2 p = ((const uint2*)(F + row * DHID))[lane];
  float fx = bfbits2f(p.x & 0xffffu), fy = bfbits2f(p.x >> 16);
  float fz = bfbits2f(p.y & 0xffffu), fw = bfbits2f(p.y >> 16);
  float4 al = ((const float4*)attn_l)[lane];
  float4 ar = ((const float4*)attn_r)[lane];
  float sl = fx*al.x + fy*al.y + fz*al.z + fw*al.w;
  float sr = fx*ar.x + fy*ar.y + fz*ar.z + fw*ar.w;
#pragma unroll
  for (int off = 16; off > 0; off >>= 1) {
    sl += __shfl_xor(sl, off, 32);
    sr += __shfl_xor(sr, off, 32);
  }
  if (lane == 0) { el[row] = sl; er[row] = sr; }
}

__global__ __launch_bounds__(256) void init_node_buffers(float* __restrict__ mmax,
                                                         float* __restrict__ ssum, int n) {
  int i = blockIdx.x * 256 + threadIdx.x;
  if (i < n) { mmax[i] = -3.402823466e38f; ssum[i] = 0.0f; }
}

// rst initialized to the broadcast GAT bias: bias-add becomes free.
__global__ __launch_bounds__(256) void init_rst_bias(const float* __restrict__ bias,
                                                     float* __restrict__ rst, long long total) {
  long long i = (long long)blockIdx.x * 256 + threadIdx.x;
  if (i < total) rst[i] = bias[i & (DHID - 1)];
}

// pass 1: e = leaky_relu(el[src]+er[dst]); segment max into mmax[dst]
__global__ __launch_bounds__(256) void edge_score(const int* __restrict__ src,
                                                  const int* __restrict__ dst,
                                                  const float* __restrict__ el,
                                                  const float* __restrict__ er,
                                                  float* __restrict__ eedge,
                                                  float* __restrict__ mmax, int e) {
  int i = blockIdx.x * 256 + threadIdx.x;
  if (i >= e) return;
  int s = src[i], d = dst[i];
  float v = el[s] + er[d];
  v = (v > 0.0f) ? v : NSLOPE * v;
  eedge[i] = v;
  atomicMaxFloat(&mmax[d], v);
}

// pass 2: ex = exp(e - m[dst]); segment sum into ssum[dst]
__global__ __launch_bounds__(256) void edge_softmax_denom(const int* __restrict__ dst,
                                                          float* __restrict__ eedge,
                                                          const float* __restrict__ mmax,
                                                          float* __restrict__ ssum, int e) {
  int i = blockIdx.x * 256 + threadIdx.x;
  if (i >= e) return;
  int d = dst[i];
  float ex = __expf(eedge[i] - mmax[d]);
  eedge[i] = ex;
  atomicAdd(&ssum[d], ex);
}

// pass 3 (bandwidth-critical): rst[dst] += alpha * feat[src]; one wave/edge.
// feat gathered as packed bf16 (8B/lane, halves gather traffic vs fp32);
// accumulation stays fp32 via global_atomic_add_f32.
__global__ __launch_bounds__(256) void edge_aggregate(const int* __restrict__ src,
                                                      const int* __restrict__ dst,
                                                      const __bf16* __restrict__ F,
                                                      const float* __restrict__ exv,
                                                      const float* __restrict__ ssum,
                                                      float* __restrict__ rst, int e) {
  long long idx = (long long)blockIdx.x * 8 + (threadIdx.x >> 5);
  if (idx >= e) return;
  int lane = threadIdx.x & 31;
  int s = src[idx], d = dst[idx];
  float alpha = exv[idx] / ssum[d];
  uint2 p = ((const uint2*)(F + (long long)s * DHID))[lane];
  float* rp = rst + (long long)d * DHID + lane * 4;
  atomicAdd(rp + 0, bfbits2f(p.x & 0xffffu) * alpha);
  atomicAdd(rp + 1, bfbits2f(p.x >> 16)     * alpha);
  atomicAdd(rp + 2, bfbits2f(p.y & 0xffffu) * alpha);
  atomicAdd(rp + 3, bfbits2f(p.y >> 16)     * alpha);
}

// ---------------------------------------------------------------------------
// Fused LSTM cell: gates = x2@W_ih^T + h0@W_hh^T + b, then gate math.
// Block = 16 rows, 8 waves; each wave owns one 16-col slice of the 128 output
// dims and accumulates its i/f/g/o tiles (4 acc tiles, 8 WMMA per K-step).
// A tiles staged in LDS as bf16 (ds path); B read from cached bf16 weights.
// Writes h1 (twice, for h1 and h[None]) and c1 directly -- gates never hit HBM.
// ---------------------------------------------------------------------------
__global__ __launch_bounds__(256) void lstm_wmma(const float* __restrict__ X2,
                                                 const float* __restrict__ H0,
                                                 const float* __restrict__ C0,
                                                 const __bf16* __restrict__ Wihb,
                                                 const __bf16* __restrict__ Whhb,
                                                 const float* __restrict__ b_ih,
                                                 const float* __restrict__ b_hh,
                                                 float* __restrict__ out,
                                                 int nrows, long long nout) {
  __shared__ __align__(32) __bf16 As[16 * DHID];
  __shared__ __align__(32) __bf16 Hs[16 * DHID];
  const long long rowBase = (long long)blockIdx.x * 16;
  if (rowBase + 16 > nrows) return;
  const int tid = threadIdx.x;
  for (int i = tid; i < 16 * DHID; i += 256) {
    As[i] = f2bf(X2[rowBase * DHID + i]);      // read x2 rows BEFORE c1 overwrite
    Hs[i] = f2bf(H0[rowBase * DHID + i]);
  }
  __syncthreads();

  const int lane = tid & 31, wave = tid >> 5;
  const int am = lane & 15, ah = lane >> 4;

  v8f acc[4];
#pragma unroll
  for (int g = 0; g < 4; ++g)
#pragma unroll
    for (int j = 0; j < 8; ++j) acc[g][j] = 0.0f;

#pragma unroll
  for (int kk = 0; kk < 4; ++kk) {
    v8bf a0 = *(const v8bf*)(As + am * DHID + kk * 32 + ah * 16);
    v8bf a1 = *(const v8bf*)(As + am * DHID + kk * 32 + ah * 16 + 8);
    v8bf hA = *(const v8bf*)(Hs + am * DHID + kk * 32 + ah * 16);
    v8bf hB = *(const v8bf*)(Hs + am * DHID + kk * 32 + ah * 16 + 8);
    v16bf a, hf;
#pragma unroll
    for (int i = 0; i < 8; ++i) { a[i] = a0[i]; a[8 + i] = a1[i]; hf[i] = hA[i]; hf[8 + i] = hB[i]; }
#pragma unroll
    for (int g = 0; g < 4; ++g) {              // i, f, g, o gate tiles
      const int j0 = g * DHID + wave * 16 + am;
      v16bf b1 = *(const v16bf*)(Wihb + (long long)j0 * DHID + kk * 32 + ah * 16);
      acc[g] = __builtin_amdgcn_wmma_f32_16x16x32_bf16(
          false, a, false, b1, (short)0, acc[g], false, false);
      v16bf b2 = *(const v16bf*)(Whhb + (long long)j0 * DHID + kk * 32 + ah * 16);
      acc[g] = __builtin_amdgcn_wmma_f32_16x16x32_bf16(
          false, hf, false, b2, (short)0, acc[g], false, false);
    }
  }

  const int col = wave * 16 + am;
#pragma unroll
  for (int g = 0; g < 8; ++g) {
    const long long row = rowBase + g + ah * 8;
    float gi = acc[0][g] + b_ih[col]            + b_hh[col];
    float gf = acc[1][g] + b_ih[DHID + col]     + b_hh[DHID + col];
    float gg = acc[2][g] + b_ih[2 * DHID + col] + b_hh[2 * DHID + col];
    float go = acc[3][g] + b_ih[3 * DHID + col] + b_hh[3 * DHID + col];
    float c0 = C0[row * DHID + col];
    float c1 = sigmoidf_(gf) * c0 + sigmoidf_(gi) * tanhf(gg);
    float h1 = sigmoidf_(go) * tanhf(c1);
    out[row * DHID + col] = h1;                  // h1
    out[nout + row * DHID + col] = h1;           // h[None]
    out[2 * nout + row * DHID + col] = c1;       // c[None]
  }
}

// ---------------------------------------------------------------------------
extern "C" void kernel_launch(void* const* d_in, const int* in_sizes, int n_in,
                              void* d_out, int out_size, void* d_ws, size_t ws_size,
                              hipStream_t stream) {
  (void)n_in; (void)out_size; (void)ws_size;
  const float* x        = (const float*)d_in[0];
  const float* h        = (const float*)d_in[1];
  const float* c        = (const float*)d_in[2];
  const int*   src0     = (const int*)d_in[3];
  const int*   dst0     = (const int*)d_in[4];
  const int*   src1     = (const int*)d_in[5];
  const int*   dst1     = (const int*)d_in[6];
  const float* W        = (const float*)d_in[7];
  const float* attn_l   = (const float*)d_in[8];
  const float* attn_r   = (const float*)d_in[9];
  const float* gat_bias = (const float*)d_in[10];
  const float* W_ih     = (const float*)d_in[11];
  const float* W_hh     = (const float*)d_in[12];
  const float* b_ih     = (const float*)d_in[13];
  const float* b_hh     = (const float*)d_in[14];

  const int n = in_sizes[0] / DHID;            // nodes
  const int e = in_sizes[3];                   // edges per block
  const long long nout = (long long)n * DHID;

  // Big node buffers live inside d_out (dead before final writes):
  //   region 0 -> bf16 feat scratch, region 2 -> fp32 rst/x1/x2 (c1 written last)
  float*  outp  = (float*)d_out;
  __bf16* featA = (__bf16*)d_out;
  float*  bufB  = outp + 2 * nout;

  size_t off = 0;
  auto alloc = [&](size_t bytes) -> void* {
    off = (off + 255) & ~(size_t)255;
    void* p = (char*)d_ws + off;
    off += bytes;
    return p;
  };
  float*  eedge = (float*)alloc(sizeof(float) * (size_t)e);
  float*  el    = (float*)alloc(sizeof(float) * (size_t)n);
  float*  er    = (float*)alloc(sizeof(float) * (size_t)n);
  float*  mmax  = (float*)alloc(sizeof(float) * (size_t)n);
  float*  ssum  = (float*)alloc(sizeof(float) * (size_t)n);
  __bf16* Wt    = (__bf16*)alloc(sizeof(__bf16) * DHID * DHID);
  __bf16* Wihb  = (__bf16*)alloc(sizeof(__bf16) * 4 * DHID * DHID);
  __bf16* Whhb  = (__bf16*)alloc(sizeof(__bf16) * 4 * DHID * DHID);

  conv_w_transpose<<<(DHID * DHID) / 256, 256, 0, stream>>>(W, Wt);
  conv_bf16<<<(4 * DHID * DHID) / 256, 256, 0, stream>>>(W_ih, Wihb, 4 * DHID * DHID);
  conv_bf16<<<(4 * DHID * DHID) / 256, 256, 0, stream>>>(W_hh, Whhb, 4 * DHID * DHID);

  for (int layer = 0; layer < 2; ++layer) {
    const float* xin = (layer == 0) ? x : bufB;
    const int* src = (layer == 0) ? src0 : src1;
    const int* dst = (layer == 0) ? dst0 : dst1;

    gat_gemm_wmma<<<(n + 63) / 64, 128, 0, stream>>>(xin, Wt, featA, n);
    node_attn_scores<<<(n + 7) / 8, 256, 0, stream>>>(featA, attn_l, attn_r, el, er, n);
    init_node_buffers<<<(n + 255) / 256, 256, 0, stream>>>(mmax, ssum, n);
    init_rst_bias<<<(int)((nout + 255) / 256), 256, 0, stream>>>(gat_bias, bufB, nout);
    edge_score<<<(e + 255) / 256, 256, 0, stream>>>(src, dst, el, er, eedge, mmax, e);
    edge_softmax_denom<<<(e + 255) / 256, 256, 0, stream>>>(dst, eedge, mmax, ssum, e);
    edge_aggregate<<<(e + 7) / 8, 256, 0, stream>>>(src, dst, featA, eedge, ssum, bufB, e);
  }

  lstm_wmma<<<(n + 15) / 16, 256, 0, stream>>>(bufB, h, c, Wihb, Whhb, b_ih, b_hh,
                                               outp, n, nout);
}
